// SPADE_Multimodal_1803886264393
// MI455X (gfx1250) — matrix operand order repaired and checked by reading, not compiled, not run on hardware
//
#include <hip/hip_runtime.h>
#include <hip/hip_bf16.h>

typedef __attribute__((ext_vector_type(16))) _Float16 v16h;
typedef __attribute__((ext_vector_type(8)))  _Float16 v8h;
typedef __attribute__((ext_vector_type(8)))  float    v8f;

#define NS     110592      // 48*48*48 voxels
#define DIM    48
#define NB     8           // batch
#define CIN    32
#define NH     64
#define NROWS  (NS / DIM)  // 2304 (b-local rows); each wave owns one row (48 = 3x16)

// ---------------------------------------------------------------------------
// Weight repack: [cls][OC][IC][27] fp32  ->  [cls][27][OC][IC] f16
// ---------------------------------------------------------------------------
__global__ void pack_w_kernel(const float* __restrict__ src,
                              _Float16* __restrict__ dst, int OC, int IC) {
  int idx = blockIdx.x * blockDim.x + threadIdx.x;
  int total = 4 * OC * IC * 27;
  if (idx >= total) return;
  int koff = idx % 27; int e = idx / 27;
  int ic = e % IC; e /= IC;
  int oc = e % OC; e /= OC;
  int cls = e;
  dst[(((size_t)cls * 27 + koff) * OC + oc) * IC + ic] = (_Float16)src[idx];
}

// ---------------------------------------------------------------------------
// InstanceNorm statistics: one block per (b,c), biased variance
// ---------------------------------------------------------------------------
__global__ void stats_kernel(const float* __restrict__ x,
                             float* __restrict__ mu, float* __restrict__ rstd) {
  int bc = blockIdx.x;                       // 0..255
  const float* xp = x + (size_t)bc * NS;
  float s = 0.f, s2 = 0.f;
  for (int i = threadIdx.x; i < NS; i += blockDim.x) {
    float v = xp[i]; s += v; s2 += v * v;
  }
  __shared__ float sh[256], sh2[256];
  sh[threadIdx.x] = s; sh2[threadIdx.x] = s2;
  __syncthreads();
  for (int off = 128; off > 0; off >>= 1) {
    if (threadIdx.x < off) {
      sh[threadIdx.x]  += sh[threadIdx.x + off];
      sh2[threadIdx.x] += sh2[threadIdx.x + off];
    }
    __syncthreads();
  }
  if (threadIdx.x == 0) {
    float m   = sh[0]  * (1.f / NS);
    float var = sh2[0] * (1.f / NS) - m * m;
    mu[bc]   = m;
    rstd[bc] = rsqrtf(var + 1e-5f);
  }
}

// ---------------------------------------------------------------------------
// x: [B][32][NS] fp32 -> xh: [B][NS][32] f16 (channel-innermost for WMMA B)
// ---------------------------------------------------------------------------
__global__ void cast_x_kernel(const float* __restrict__ x,
                              _Float16* __restrict__ xh) {
  int idx = blockIdx.x * blockDim.x + threadIdx.x;   // b*NS + s
  if (idx >= NB * NS) return;
  int b = idx / NS, s = idx % NS;
  const float* xp = x + (size_t)b * CIN * NS + s;
  _Float16* o = xh + (size_t)idx * CIN;
  #pragma unroll
  for (int c = 0; c < CIN; ++c) o[c] = (_Float16)xp[(size_t)c * NS];
}

// ---------------------------------------------------------------------------
// conv1: actv = relu(conv3d(x, Wshared) + b).  Implicit GEMM, M=64, K=27*32.
// One wave owns a 16(oc) x 48(full x-row) tile: 3 accumulators, A fragment
// loaded once per K-step and reused across 3 WMMAs.
// ---------------------------------------------------------------------------
__global__ void __launch_bounds__(256)
conv1_kernel(const _Float16* __restrict__ xh,      // [B][NS][32]
             const _Float16* __restrict__ Wp,      // [4][27][64][32]
             const float*    __restrict__ bshared, // [4][64]
             const int*      __restrict__ y,
             _Float16*       __restrict__ actv) {  // [B][NS][64]
  const int lane = threadIdx.x & 31;
  int wave = (blockIdx.x * blockDim.x + threadIdx.x) >> 5;
  const int row = wave % NROWS; wave /= NROWS;     // (d,h) row, 48 voxels
  const int mt  = wave % 4;     wave /= 4;
  const int b   = wave;
  const int cls = y[b];

  const int h = row % DIM;
  const int d = row / DIM;
  const int s0 = row * DIM;                        // first voxel of the row
  const int oc0 = mt * 16;

  const int  ncol   = lane & 15;
  const bool hi     = lane >= 16;
  const int  kbaseA = hi ? 16 : 0;   // A-frag: lanes>=16 hold K 16-31
  const int  kbaseB = hi ? 16 : 0;   // B-frag: lanes>=16 hold K 16-31

  const _Float16* xb    = xh + (size_t)b * NS * CIN;
  const _Float16* wbase = Wp + (size_t)cls * 27 * NH * CIN
                             + (size_t)(oc0 + ncol) * CIN;   // A row m = lane&15
  v8f acc0 = {}, acc1 = {}, acc2 = {};
  for (int koff = 0; koff < 27; ++koff) {
    const int kd = koff / 9, kh = (koff / 3) % 3, kw = koff % 3;
    const int dd = d + kd - 1, hh = h + kh - 1;               // wave-uniform
    if ((unsigned)dd >= (unsigned)DIM || (unsigned)hh >= (unsigned)DIM) continue;

    // A fragment: one weight tap, reused for all 3 N-subtiles
    const _Float16* wr = wbase + (size_t)koff * NH * CIN;
    v8h a0 = *(const v8h*)(wr + kbaseA);
    v8h a1 = *(const v8h*)(wr + kbaseA + 8);
    v16h afrag = __builtin_shufflevector(a0, a1,
        0,1,2,3,4,5,6,7,8,9,10,11,12,13,14,15);

    const _Float16* rowp = xb + (size_t)((dd * DIM + hh) * DIM) * CIN + kbaseB;
    #pragma unroll
    for (int sub = 0; sub < 3; ++sub) {
      const int ww = sub * 16 + ncol + kw - 1;     // only edges ever OOB
      v16h bfrag = {};
      if ((unsigned)ww < (unsigned)DIM)
        bfrag = *(const v16h*)(rowp + (size_t)ww * CIN);
      v8f* acc = (sub == 0) ? &acc0 : (sub == 1) ? &acc1 : &acc2;
      *acc = __builtin_amdgcn_wmma_f32_16x16x32_f16(
          false, afrag, false, bfrag, (short)0, *acc, false, false);
    }
  }
  // C layout: VGPR r -> M = (hi?8:0)+r, N = ncol.  oc contiguous in actv.
  const int mbase = hi ? 8 : 0;
  const float* bias = bshared + cls * NH + oc0 + mbase;
  #pragma unroll
  for (int sub = 0; sub < 3; ++sub) {
    const v8f* acc = (sub == 0) ? &acc0 : (sub == 1) ? &acc1 : &acc2;
    _Float16* o = actv + ((size_t)b * NS + (s0 + sub * 16 + ncol)) * NH
                       + oc0 + mbase;
    v8h res;
    #pragma unroll
    for (int r = 0; r < 8; ++r) {
      float v = (*acc)[r] + bias[r];
      res[r] = (_Float16)(v > 0.f ? v : 0.f);
    }
    *(v8h*)o = res;
  }
}

// ---------------------------------------------------------------------------
// Fused gamma/beta convs + instance-norm epilogue.  M=32, K=27*64.
// One wave owns 16(oc) x 48(row); each B fragment feeds 2 WMMAs (gamma+beta),
// each A fragment pair feeds 6 WMMAs.  out = norm*(1+gamma)+beta.
// ---------------------------------------------------------------------------
__global__ void __launch_bounds__(256)
conv_final_kernel(const float*    __restrict__ x,     // [B][32][NS]
                  const _Float16* __restrict__ actv,  // [B][NS][64]
                  const _Float16* __restrict__ Wpg,   // [4][27][32][64]
                  const _Float16* __restrict__ Wpb,   // [4][27][32][64]
                  const float*    __restrict__ bgamma,// [4][32]
                  const float*    __restrict__ bbeta, // [4][32]
                  const float*    __restrict__ mu,    // [B*32]
                  const float*    __restrict__ rstd,  // [B*32]
                  const int*      __restrict__ y,
                  float*          __restrict__ out) { // [B][32][NS]
  const int lane = threadIdx.x & 31;
  int wave = (blockIdx.x * blockDim.x + threadIdx.x) >> 5;
  const int row = wave % NROWS; wave /= NROWS;
  const int mt  = wave % 2;     wave /= 2;
  const int b   = wave;
  const int cls = y[b];

  const int h = row % DIM;
  const int d = row / DIM;
  const int s0 = row * DIM;
  const int oc0 = mt * 16;

  const int  ncol   = lane & 15;
  const bool hi     = lane >= 16;
  const int  kbaseA = hi ? 16 : 0;
  const int  kbaseB = hi ? 16 : 0;

  const _Float16* ab     = actv + (size_t)b * NS * NH;
  const _Float16* wgbase = Wpg + (size_t)cls * 27 * CIN * NH
                               + (size_t)(oc0 + ncol) * NH;
  const _Float16* wbbase = Wpb + (size_t)cls * 27 * CIN * NH
                               + (size_t)(oc0 + ncol) * NH;
  v8f ag0 = {}, ag1 = {}, ag2 = {};      // gamma accumulators per subtile
  v8f ab0 = {}, ab1 = {}, ab2 = {};      // beta accumulators per subtile
  for (int koff = 0; koff < 27; ++koff) {
    const int kd = koff / 9, kh = (koff / 3) % 3, kw = koff % 3;
    const int dd = d + kd - 1, hh = h + kh - 1;    // wave-uniform
    if ((unsigned)dd >= (unsigned)DIM || (unsigned)hh >= (unsigned)DIM) continue;
    const _Float16* rowp = ab + (size_t)((dd * DIM + hh) * DIM) * NH + kbaseB;
    #pragma unroll
    for (int half = 0; half < 2; ++half) {         // NH=64 -> two K=32 steps
      const int ic0 = half * 32;
      const size_t woff = (size_t)koff * CIN * NH + ic0 + kbaseA;
      v8h g0 = *(const v8h*)(wgbase + woff);
      v8h g1 = *(const v8h*)(wgbase + woff + 8);
      v16h afg = __builtin_shufflevector(g0, g1,
          0,1,2,3,4,5,6,7,8,9,10,11,12,13,14,15);
      v8h t0 = *(const v8h*)(wbbase + woff);
      v8h t1 = *(const v8h*)(wbbase + woff + 8);
      v16h afb = __builtin_shufflevector(t0, t1,
          0,1,2,3,4,5,6,7,8,9,10,11,12,13,14,15);
      #pragma unroll
      for (int sub = 0; sub < 3; ++sub) {
        const int ww = sub * 16 + ncol + kw - 1;
        v16h bfrag = {};
        if ((unsigned)ww < (unsigned)DIM)
          bfrag = *(const v16h*)(rowp + (size_t)ww * NH + ic0);
        v8f* accg = (sub == 0) ? &ag0 : (sub == 1) ? &ag1 : &ag2;
        v8f* accb = (sub == 0) ? &ab0 : (sub == 1) ? &ab1 : &ab2;
        *accg = __builtin_amdgcn_wmma_f32_16x16x32_f16(
            false, afg, false, bfrag, (short)0, *accg, false, false);
        *accb = __builtin_amdgcn_wmma_f32_16x16x32_f16(
            false, afb, false, bfrag, (short)0, *accb, false, false);
      }
    }
  }
  // Epilogue: instance-norm + SPADE modulation, fused.
  const int mbase = hi ? 8 : 0;
  #pragma unroll
  for (int sub = 0; sub < 3; ++sub) {
    const v8f* accg = (sub == 0) ? &ag0 : (sub == 1) ? &ag1 : &ag2;
    const v8f* accb = (sub == 0) ? &ab0 : (sub == 1) ? &ab1 : &ab2;
    const int s = s0 + sub * 16 + ncol;
    #pragma unroll
    for (int r = 0; r < 8; ++r) {
      const int oc = oc0 + mbase + r;
      const float g  = (*accg)[r] + bgamma[cls * CIN + oc];
      const float be = (*accb)[r] + bbeta [cls * CIN + oc];
      const size_t xi = ((size_t)b * CIN + oc) * NS + s;
      const float norm = (x[xi] - mu[b * CIN + oc]) * rstd[b * CIN + oc];
      out[xi] = norm * (1.f + g) + be;
    }
  }
}

// ---------------------------------------------------------------------------
extern "C" void kernel_launch(void* const* d_in, const int* in_sizes, int n_in,
                              void* d_out, int out_size, void* d_ws, size_t ws_size,
                              hipStream_t stream) {
  const float* x       = (const float*)d_in[0];
  const int*   y       = (const int*)  d_in[1];
  const float* Wshared = (const float*)d_in[2];
  const float* bshared = (const float*)d_in[3];
  const float* Wgamma  = (const float*)d_in[4];
  const float* bgamma  = (const float*)d_in[5];
  const float* Wbeta   = (const float*)d_in[6];
  const float* bbeta   = (const float*)d_in[7];
  float* out = (float*)d_out;

  char* ws = (char*)d_ws;              // all offsets 256B-aligned
  float*    mu   = (float*)   (ws + 0);          //  1 KB
  float*    rstd = (float*)   (ws + 1024);       //  1 KB
  _Float16* Wp1  = (_Float16*)(ws + 2048);       //  442368 B [4][27][64][32]
  _Float16* Wpg  = (_Float16*)(ws + 444416);     //  442368 B [4][27][32][64]
  _Float16* Wpb  = (_Float16*)(ws + 886784);     //  442368 B
  _Float16* xh   = (_Float16*)(ws + 1329152);    //  56.6 MB [8][NS][32]
  _Float16* actv = (_Float16*)(ws + 57952256);   // 113.2 MB [8][NS][64]

  pack_w_kernel<<<864, 256, 0, stream>>>(Wshared, Wp1, NH, CIN);
  pack_w_kernel<<<864, 256, 0, stream>>>(Wgamma,  Wpg, CIN, NH);
  pack_w_kernel<<<864, 256, 0, stream>>>(Wbeta,   Wpb, CIN, NH);
  stats_kernel <<<NB * CIN, 256, 0, stream>>>(x, mu, rstd);
  cast_x_kernel<<<(NB * NS) / 256, 256, 0, stream>>>(x, xh);
  // conv1: 8 * 4 * 2304 waves = 73728; * 32 / 256 = 9216 blocks (exact)
  conv1_kernel<<<9216, 256, 0, stream>>>(xh, Wp1, bshared, y, actv);
  // final: 8 * 2 * 2304 waves = 36864; * 32 / 256 = 4608 blocks (exact)
  conv_final_kernel<<<4608, 256, 0, stream>>>(x, actv, Wpg, Wpb,
                                              bgamma, bbeta, mu, rstd, y, out);
}
